// HPNS_RNNAgent_45689862095548
// MI455X (gfx1250) — matrix operand order repaired and checked by reading, not compiled, not run on hardware
//
#include <hip/hip_runtime.h>
#include <hip/hip_bf16.h>
#include <math.h>

// ---------------------------------------------------------------------------
// HPNS_RNNAgent fused pipeline for gfx1250 (wave32, WMMA f32_16x16x32_f16)
//
//   emb[r, o] = [ef⊗relu1 | ef][r, 0..671] @ W'[672, 256]   (bias folded in K)
//
// hyper_gemm v2: 8 agents (M=128) per 256-thread WG, 8 waves x N=32.
//   - W' read once per WG (88 MB total L2 traffic instead of 700 MB)
//   - A fragments synthesized on the fly: within a 32-wide k-step the feature
//     index i = kb/64 is constant, so A-frag = splat(ef[m][i]) * relu1-frag.
// ---------------------------------------------------------------------------

typedef __attribute__((ext_vector_type(16))) _Float16 v16h;
typedef __attribute__((ext_vector_type(8)))  float    v8f;

constexpr int RAG  = 2048;   // BS*NA agents
constexpr int NALY = 15;
constexpr int HDIM = 64;     // H
constexpr int NOUT = 256;    // HD*H
constexpr int KP   = 672;    // 640 (=10*64) + 10 bias cols + pad to 21*32
constexpr int GA   = 8;      // agents per workgroup in hyper_gemm
constexpr int RSTR = 80;     // LDS stride (halves) for relu1 rows (16B aligned, ~2-way banks)
constexpr int ESTR = 48;     // LDS stride (halves) for ef rows

static __device__ __forceinline__ v16h mk_frag(uint4 a, uint4 b) {
  v16h r;
  __builtin_memcpy(&r, &a, 16);
  __builtin_memcpy(reinterpret_cast<char*>(&r) + 16, &b, 16);
  return r;
}

static __device__ __forceinline__ v8f wmma16(v16h a, v16h b, v8f c) {
  return __builtin_amdgcn_wmma_f32_16x16x32_f16(false, a, false, b,
                                                (short)0, c, false, false);
}

// --- weight reordering: W'[n][j], j=(i*64+k) -> w2[k][i*256+n]; j=640+i -> b2[i*256+n]
__global__ void prep_weights(const float* __restrict__ he_w2, const float* __restrict__ he_b2,
                             const float* __restrict__ ha_w2, const float* __restrict__ ha_b2,
                             const float* __restrict__ pass_w1,
                             _Float16* __restrict__ WEp, _Float16* __restrict__ WAp,
                             _Float16* __restrict__ pw1h) {
  const int WSZ = NOUT * KP;  // 172032
  int idx = blockIdx.x * 256 + threadIdx.x;
  if (idx < WSZ) {
    int n = idx / KP, j = idx - n * KP;
    float v = 0.f;
    if (j < 640)      v = he_w2[(j & 63) * 2820 + (j >> 6) * NOUT + n];
    else if (j < 650) v = he_b2[(j - 640) * NOUT + n];
    WEp[idx] = (_Float16)v;
  } else if (idx < 2 * WSZ) {
    int i2 = idx - WSZ;
    int n = i2 / KP, j = i2 - n * KP;
    float v = 0.f;
    if (j < 640)      v = ha_w2[(j & 63) * 2560 + (j >> 6) * NOUT + n];
    else if (j < 650) v = ha_b2[(j - 640) * NOUT + n];
    WAp[i2] = (_Float16)v;
  } else if (idx < 2 * WSZ + HDIM * HDIM) {
    int i2 = idx - 2 * WSZ;
    int n = i2 >> 6, k = i2 & 63;
    pw1h[i2] = (_Float16)pass_w1[k * HDIM + n];  // N-major for B fragments
  }
}

// --- hypernet layer 1 (K=10, tiny): relu(feats@W + b) in f16
__global__ __launch_bounds__(64)
void relu_layer(const float* __restrict__ feats, const float* __restrict__ W,
                const float* __restrict__ b, _Float16* __restrict__ out, int indim) {
  int r = blockIdx.x, c = threadIdx.x;
  float s = b[c];
  for (int i = 0; i < indim; ++i) s += feats[r * indim + i] * W[i * HDIM + c];
  out[r * HDIM + c] = (_Float16)fmaxf(s, 0.f);
}

// --- own embedding (linear, f32, K=30)
__global__ __launch_bounds__(64)
void own_layer(const float* __restrict__ feats, const float* __restrict__ W,
               const float* __restrict__ b, float* __restrict__ out) {
  int r = blockIdx.x, c = threadIdx.x;
  float s = b[c];
  for (int i = 0; i < 30; ++i) s += feats[r * 30 + i] * W[i * HDIM + c];
  out[r * HDIM + c] = s;
}

// --- main GEMM: M=128 (8 agents) per WG, B = W'[256][672] f16 streamed from L2
//     once per WG, A-frags synthesized from LDS-staged ef / relu1 rows.
__global__ __launch_bounds__(256)
void hyper_gemm(const float* __restrict__ feats, const _Float16* __restrict__ relu1,
                const _Float16* __restrict__ Wp, float* __restrict__ sumOut,
                _Float16* __restrict__ rowOut, int perAgent) {
  __shared__ _Float16 rl[GA * 16 * RSTR];  // relu1 rows, padded stride   (20 KB)
  __shared__ _Float16 ep[GA * 16 * ESTR];  // ef rows: 10 vals + zero pad (12 KB)
  const int tid = threadIdx.x;             // 0..255, 8 waves
  const int wave = tid >> 5, lane = tid & 31;
  const int mrow = lane & 15, hi = lane >> 4;
  const int agent0 = blockIdx.x * GA;

  // Stage relu1 (f16) and ef (f32->f16, zero-padded to 32) rows; pad rows -> 0.
  for (int idx = tid; idx < GA * 16 * 64; idx += 256) {
    int row = idx >> 6, k = idx & 63;
    int ag = row >> 4, m = row & 15;
    _Float16 v = (_Float16)0.f;
    if (m < perAgent) v = relu1[((agent0 + ag) * perAgent + m) * HDIM + k];
    rl[row * RSTR + k] = v;
  }
  for (int idx = tid; idx < GA * 16 * 32; idx += 256) {
    int row = idx >> 5, j = idx & 31;
    int ag = row >> 4, m = row & 15;
    _Float16 v = (_Float16)0.f;
    if (m < perAgent && j < 10)
      v = (_Float16)feats[((agent0 + ag) * perAgent + m) * 10 + j];
    ep[row * ESTR + j] = v;
  }
  __syncthreads();

  const int n0 = wave * 32;                 // each wave owns 32 of 256 cols
  v8f acc[GA][2] = {};

  // Main k-steps 0..19: i = kk/2 constant per step -> A-frag = ef_i * relu1-frag.
  for (int kk = 0; kk < 20; ++kk) {
    const int kb = kk * 32;
    const int i = kk >> 1;
    const int kbase = (kk & 1) * 32;
    const _Float16* bp0 = Wp + (size_t)(n0 + mrow) * KP + kb + hi * 8;
    const _Float16* bp1 = Wp + (size_t)(n0 + 16 + mrow) * KP + kb + hi * 8;
    v16h bf0 = mk_frag(*(const uint4*)bp0, *(const uint4*)(bp0 + 16));
    v16h bf1 = mk_frag(*(const uint4*)bp1, *(const uint4*)(bp1 + 16));
#pragma unroll
    for (int t = 0; t < GA; ++t) {
      const int row = t * 16 + mrow;
      const _Float16* rp = &rl[row * RSTR + kbase + hi * 8];
      v16h rf = mk_frag(*(const uint4*)rp, *(const uint4*)(rp + 16));
      _Float16 efs = ep[row * ESTR + i];
      v16h af = rf * efs;                    // v_pk_mul_f16 splat
      acc[t][0] = wmma16(af, bf0, acc[t][0]);
      acc[t][1] = wmma16(af, bf1, acc[t][1]);
    }
  }
  // Bias k-step (kk = 20): A-frag is the zero-padded ef row itself.
  {
    const int kb = 640;
    const _Float16* bp0 = Wp + (size_t)(n0 + mrow) * KP + kb + hi * 8;
    const _Float16* bp1 = Wp + (size_t)(n0 + 16 + mrow) * KP + kb + hi * 8;
    v16h bf0 = mk_frag(*(const uint4*)bp0, *(const uint4*)(bp0 + 16));
    v16h bf1 = mk_frag(*(const uint4*)bp1, *(const uint4*)(bp1 + 16));
#pragma unroll
    for (int t = 0; t < GA; ++t) {
      const _Float16* apx = &ep[(t * 16 + mrow) * ESTR + hi * 8];
      v16h af = mk_frag(*(const uint4*)apx, *(const uint4*)(apx + 16));
      acc[t][0] = wmma16(af, bf0, acc[t][0]);
      acc[t][1] = wmma16(af, bf1, acc[t][1]);
    }
  }

  if (rowOut) {  // ally path: keep per-ally rows (f16) for the passing MLP
#pragma unroll
    for (int t = 0; t < GA; ++t)
#pragma unroll
      for (int u = 0; u < 2; ++u)
#pragma unroll
        for (int v = 0; v < 8; ++v)
          rowOut[(size_t)((agent0 + t) * 16 + v + hi * 8) * NOUT +
                 n0 + u * 16 + mrow] = (_Float16)acc[t][u][v];
  }
  // Sum over the 16 rows (enemies / allies+pad): 8 VGPRs + lane-half combine.
#pragma unroll
  for (int t = 0; t < GA; ++t)
#pragma unroll
    for (int u = 0; u < 2; ++u) {
      float s = 0.f;
#pragma unroll
      for (int v = 0; v < 8; ++v) s += acc[t][u][v];
      s += __shfl_xor(s, 16);
      if (hi == 0) sumOut[(agent0 + t) * NOUT + n0 + u * 16 + mrow] = s;
    }
}

// --- passing MLP: X1 = relu(emb_al4h @ pw1 + b1) via WMMA, then N=3 layer +
//     butterfly row-reduction in-register.
__global__ __launch_bounds__(128)
void pass_gemm(const _Float16* __restrict__ X, const _Float16* __restrict__ pw1h,
               const float* __restrict__ b1, const float* __restrict__ w2,
               const float* __restrict__ b2, float* __restrict__ pout) {
  const int tid = threadIdx.x;
  const int wave = tid >> 5, lane = tid & 31;
  const int mrow = lane & 15, hi = lane >> 4;
  const int m0 = blockIdx.x * 64 + wave * 16;

  v8f acc[4] = {};
#pragma unroll
  for (int kk = 0; kk < 2; ++kk) {
    int kb = kk * 32;
    const _Float16* ar = X + (size_t)(m0 + mrow) * HDIM + kb + hi * 8;
    uint4 a0 = *reinterpret_cast<const uint4*>(ar);
    uint4 a1 = *reinterpret_cast<const uint4*>(ar + 16);
    v16h af = mk_frag(a0, a1);
#pragma unroll
    for (int t = 0; t < 4; ++t) {
      const _Float16* bp = pw1h + (t * 16 + mrow) * HDIM + kb + hi * 8;
      uint4 b0 = *reinterpret_cast<const uint4*>(bp);
      uint4 b1 = *reinterpret_cast<const uint4*>(bp + 16);
      acc[t] = wmma16(af, mk_frag(b0, b1), acc[t]);
    }
  }

  float bb[4], w2r[4][3];
#pragma unroll
  for (int t = 0; t < 4; ++t) {
    int n = t * 16 + mrow;
    bb[t] = b1[n];
#pragma unroll
    for (int j = 0; j < 3; ++j) w2r[t][j] = w2[n * 3 + j];
  }
#pragma unroll
  for (int v = 0; v < 8; ++v) {
    float p[3] = {0.f, 0.f, 0.f};
#pragma unroll
    for (int t = 0; t < 4; ++t) {
      float x = fmaxf(acc[t][v] + bb[t], 0.f);
#pragma unroll
      for (int j = 0; j < 3; ++j) p[j] += x * w2r[t][j];
    }
#pragma unroll
    for (int j = 0; j < 3; ++j) {
      p[j] += __shfl_xor(p[j], 1);
      p[j] += __shfl_xor(p[j], 2);
      p[j] += __shfl_xor(p[j], 4);
      p[j] += __shfl_xor(p[j], 8);
    }
    if (mrow == 0) {
      int arow = m0 + v + hi * 8;
#pragma unroll
      for (int j = 0; j < 3; ++j) pout[arow * 3 + j] = p[j] + b2[j];
    }
  }
}

// --- fused tail: passing-embed max/mean, softmax merger, GRU cell, fc2 head, q assembly.
__global__ __launch_bounds__(64)
void final_stage(const float* __restrict__ en_sum, const float* __restrict__ al_sum,
                 const float* __restrict__ emb_own, const float* __restrict__ hidden,
                 const float* __restrict__ merger_w,
                 const float* __restrict__ wih, const float* __restrict__ whh,
                 const float* __restrict__ bih, const float* __restrict__ bhh,
                 const float* __restrict__ fw1, const float* __restrict__ fb1,
                 const float* __restrict__ fw2, const float* __restrict__ fb2,
                 const float* __restrict__ pout,
                 float* __restrict__ out_q, float* __restrict__ out_h) {
  __shared__ float xs[HDIM], hs[HDIM], hhs[HDIM], y1s[HDIM], pes[3];
  const int agent = blockIdx.x;
  const int h = threadIdx.x;

  if (h < 3) {  // mean over HD, max over real allies
    float mx = -1e30f;
    for (int a = 0; a < NALY; ++a) {
      float s = 0.f;
      int base = (agent * 16 + a) * 4;
      for (int d = 0; d < 4; ++d) s += pout[(base + d) * 3 + h];
      mx = fmaxf(mx, 0.25f * s);
    }
    pes[h] = mx;
  }

  float e[4], esum = 0.f;
  for (int d = 0; d < 4; ++d) { e[d] = __expf(merger_w[d * HDIM + h]); esum += e[d]; }
  float merged = 0.f;
  for (int d = 0; d < 4; ++d)
    merged += (e[d] / esum) *
              (en_sum[agent * NOUT + d * HDIM + h] + al_sum[agent * NOUT + d * HDIM + h]);
  xs[h] = fmaxf(emb_own[agent * HDIM + h] + merged, 0.f);
  float hin = hidden[agent * HDIM + h];
  hs[h] = hin;
  __syncthreads();

  float gi[3], gh[3];
  for (int g = 0; g < 3; ++g) {
    int c = g * HDIM + h;
    float si = bih[c], sh = bhh[c];
    for (int k = 0; k < HDIM; ++k) {
      si += xs[k] * wih[c * HDIM + k];
      sh += hs[k] * whh[c * HDIM + k];
    }
    gi[g] = si; gh[g] = sh;
  }
  float rg = 1.f / (1.f + __expf(-(gi[0] + gh[0])));
  float zg = 1.f / (1.f + __expf(-(gi[1] + gh[1])));
  float ng = tanhf(gi[2] + rg * gh[2]);
  float hh = (1.f - zg) * ng + zg * hin;
  hhs[h] = hh;
  out_h[agent * HDIM + h] = hh;
  __syncthreads();

  float y = fb1[h];
  for (int k = 0; k < HDIM; ++k) y += hhs[k] * fw1[k * HDIM + h];
  y1s[h] = fmaxf(y, 0.f);
  __syncthreads();

  if (h < 19) {
    float q = fb2[h];
    for (int k = 0; k < HDIM; ++k) q += y1s[k] * fw2[k * 19 + h];
    out_q[agent * 22 + (h < 9 ? h : h + 3)] = q;
  } else if (h < 22) {
    out_q[agent * 22 + 9 + (h - 19)] = pes[h - 19];
  }
}

extern "C" void kernel_launch(void* const* d_in, const int* in_sizes, int n_in,
                              void* d_out, int out_size, void* d_ws, size_t ws_size,
                              hipStream_t stream) {
  const float* own_feats   = (const float*)d_in[1];
  const float* ally_feats  = (const float*)d_in[2];
  const float* enemy_feats = (const float*)d_in[3];
  const float* hidden      = (const float*)d_in[4];
  const float* fc1_w = (const float*)d_in[5];
  const float* fc1_b = (const float*)d_in[6];
  const float* he_w1 = (const float*)d_in[7];
  const float* he_b1 = (const float*)d_in[8];
  const float* he_w2 = (const float*)d_in[9];
  const float* he_b2 = (const float*)d_in[10];
  const float* ha_w1 = (const float*)d_in[11];
  const float* ha_b1 = (const float*)d_in[12];
  const float* ha_w2 = (const float*)d_in[13];
  const float* ha_b2 = (const float*)d_in[14];
  const float* merger_w = (const float*)d_in[15];
  const float* gru_wih = (const float*)d_in[16];
  const float* gru_whh = (const float*)d_in[17];
  const float* gru_bih = (const float*)d_in[18];
  const float* gru_bhh = (const float*)d_in[19];
  const float* fc2_w1 = (const float*)d_in[20];
  const float* fc2_b1 = (const float*)d_in[21];
  const float* fc2_w2 = (const float*)d_in[22];
  const float* fc2_b2 = (const float*)d_in[23];
  const float* pass_w1 = (const float*)d_in[24];
  const float* pass_b1 = (const float*)d_in[25];
  const float* pass_w2 = (const float*)d_in[26];
  const float* pass_b2 = (const float*)d_in[27];

  char* w = (char*)d_ws;
  auto alloc = [&](size_t bytes) {
    char* p = w;
    w += (bytes + 255) & ~size_t(255);
    return p;
  };
  _Float16* WEp    = (_Float16*)alloc((size_t)NOUT * KP * 2);
  _Float16* WAp    = (_Float16*)alloc((size_t)NOUT * KP * 2);
  _Float16* pw1h   = (_Float16*)alloc((size_t)HDIM * HDIM * 2);
  _Float16* relu1e = (_Float16*)alloc((size_t)32768 * HDIM * 2);
  _Float16* relu1a = (_Float16*)alloc((size_t)30720 * HDIM * 2);
  float*    embown = (float*)alloc((size_t)RAG * HDIM * 4);
  float*    en_sum = (float*)alloc((size_t)RAG * NOUT * 4);
  float*    al_sum = (float*)alloc((size_t)RAG * NOUT * 4);
  _Float16* al4h   = (_Float16*)alloc((size_t)32768 * NOUT * 2);
  float*    pout   = (float*)alloc((size_t)131072 * 3 * 4);

  float* out_q = (float*)d_out;        // [2048][22]
  float* out_h = out_q + RAG * 22;     // [2048][64]

  prep_weights<<<1360, 256, 0, stream>>>(he_w2, he_b2, ha_w2, ha_b2, pass_w1,
                                         WEp, WAp, pw1h);
  relu_layer<<<32768, 64, 0, stream>>>(enemy_feats, he_w1, he_b1, relu1e, 10);
  relu_layer<<<30720, 64, 0, stream>>>(ally_feats, ha_w1, ha_b1, relu1a, 10);
  own_layer<<<RAG, 64, 0, stream>>>(own_feats, fc1_w, fc1_b, embown);
  hyper_gemm<<<RAG / GA, 256, 0, stream>>>(enemy_feats, relu1e, WEp, en_sum,
                                           (_Float16*)nullptr, 16);
  hyper_gemm<<<RAG / GA, 256, 0, stream>>>(ally_feats, relu1a, WAp, al_sum,
                                           al4h, 15);
  pass_gemm<<<RAG, 128, 0, stream>>>(al4h, pw1h, pass_b1, pass_w2, pass_b2, pout);
  final_stage<<<RAG, 64, 0, stream>>>(en_sum, al_sum, embown, hidden, merger_w,
                                      gru_wih, gru_whh, gru_bih, gru_bhh,
                                      fc2_w1, fc2_b1, fc2_w2, fc2_b2, pout,
                                      out_q, out_h);
}
